// SentenceLlamaAttention_67740224192514
// MI455X (gfx1250) — matrix operand order repaired and checked by reading, not compiled, not run on hardware
//
#include <hip/hip_runtime.h>
#include <hip/hip_bf16.h>

typedef __attribute__((ext_vector_type(16))) _Float16 v16h;
typedef __attribute__((ext_vector_type(8)))  float    v8f;

#define S_LEN 4096
#define DMODEL 1024
#define NHEAD 16
#define NKVH 4
#define HDIM 64

union F16x16 {
    v16h v;
    _Float16 h[16];
    unsigned int u[8];
};

__device__ __forceinline__ unsigned int pack2(_Float16 a, _Float16 b) {
    union { _Float16 h[2]; unsigned int u; } x;
    x.h[0] = a; x.h[1] = b;
    return x.u;
}

// ---------------------------------------------------------------------------
// Generic tiled GEMM: C[M x N] = A[M x K] * B[K x N]
//   A fp32 or f16 (template), B fp32, C f16 (ws) or fp32 (out)
//   block = 256 threads (8 waves), tile 128(M) x 128(N), BK = 32
//   per wave per iter: one A-frag, 8 B-frags -> 8 x v_wmma_f32_16x16x32_f16
// ---------------------------------------------------------------------------
template<bool A_IS_F16, bool OUT_F16>
__global__ __launch_bounds__(256)
void gemm_f16wmma_kernel(const void* __restrict__ Aptr,
                         const float* __restrict__ B,
                         void* __restrict__ Cptr,
                         int M, int N, int K)
{
    __shared__ _Float16 Al[128 * 32];    // [row][k]
    __shared__ _Float16 Btl[128 * 32];   // [n][k]  (B transposed)

    const int t    = threadIdx.x;
    const int lane = t & 31;
    const int wave = t >> 5;
    const int nl   = lane & 15;
    const int hi   = lane >> 4;
    const int mb   = blockIdx.x * 128;
    const int nb   = blockIdx.y * 128;

    v8f acc[8] = {};

    for (int kc = 0; kc < K; kc += 32) {
        // ---- stage A tile (128 x 32) as f16 ----
        if constexpr (A_IS_F16) {
            const _Float16* Af = (const _Float16*)Aptr;
            #pragma unroll
            for (int i = 0; i < 2; ++i) {
                int c    = t + 256 * i;        // 512 chunks of 8 halves
                int row  = c >> 2;
                int col8 = (c & 3) << 3;
                uint4 d = *(const uint4*)&Af[(size_t)(mb + row) * K + kc + col8];
                *(uint4*)&Al[row * 32 + col8] = d;
            }
        } else {
            const float* Af = (const float*)Aptr;
            #pragma unroll
            for (int i = 0; i < 4; ++i) {
                int c    = t + 256 * i;        // 1024 chunks of 4 floats
                int row  = c >> 3;
                int col4 = (c & 7) << 2;
                float4 f = *(const float4*)&Af[(size_t)(mb + row) * K + kc + col4];
                uint2 pk;
                pk.x = pack2((_Float16)f.x, (_Float16)f.y);
                pk.y = pack2((_Float16)f.z, (_Float16)f.w);
                *(uint2*)&Al[row * 32 + col4] = pk;
            }
        }
        // ---- stage B tile (32 x 128) transposed -> Btl[n][k] ----
        #pragma unroll
        for (int i = 0; i < 4; ++i) {
            int c    = t + 256 * i;            // 1024 chunks of 4 floats
            int krow = c >> 5;
            int col4 = (c & 31) << 2;
            float4 f = *(const float4*)&B[(size_t)(kc + krow) * N + nb + col4];
            Btl[(col4 + 0) * 32 + krow] = (_Float16)f.x;
            Btl[(col4 + 1) * 32 + krow] = (_Float16)f.y;
            Btl[(col4 + 2) * 32 + krow] = (_Float16)f.z;
            Btl[(col4 + 3) * 32 + krow] = (_Float16)f.w;
        }
        __syncthreads();

        // ---- A fragment (16 x 32), documented 16-bit A layout ----
        F16x16 af;
        {
            int row = (wave << 4) + nl;
            #pragma unroll
            for (int g = 0; g < 8; ++g) {
                int k = ((g & 4) << 2) + (hi << 3) + ((g & 3) << 1);
                af.u[g] = *(const unsigned int*)&Al[row * 32 + k];
            }
        }
        // ---- 8 B fragments + 8 WMMAs, in two preloaded groups of 4 ----
        #pragma unroll
        for (int grp = 0; grp < 2; ++grp) {
            F16x16 bf[4];
            #pragma unroll
            for (int j = 0; j < 4; ++j) {
                int n = ((grp * 4 + j) << 4) + nl;
                #pragma unroll
                for (int g = 0; g < 8; ++g)
                    bf[j].u[g] = *(const unsigned int*)&Btl[n * 32 + ((g & 4) << 2) + (hi << 3) + ((g & 3) << 1)];
            }
            #pragma unroll
            for (int j = 0; j < 4; ++j)
                acc[grp * 4 + j] = __builtin_amdgcn_wmma_f32_16x16x32_f16(
                    false, af.v, false, bf[j].v, (short)0, acc[grp * 4 + j], false, false);
        }
        __syncthreads();
    }

    // ---- epilogue: C-tile layout M = r + 8*hi, N = nl ----
    #pragma unroll
    for (int nt = 0; nt < 8; ++nt) {
        #pragma unroll
        for (int r = 0; r < 8; ++r) {
            int m = mb + (wave << 4) + r + (hi << 3);
            int n = nb + (nt << 4) + nl;
            if constexpr (OUT_F16)
                ((_Float16*)Cptr)[(size_t)m * N + n] = (_Float16)acc[nt][r];
            else
                ((float*)Cptr)[(size_t)m * N + n] = acc[nt][r];
        }
    }
}

// ---------------------------------------------------------------------------
// In-place RoPE on f16 Q [S][H*64] and K [S][KVH*64].
// ---------------------------------------------------------------------------
__global__ __launch_bounds__(256)
void rope_kernel(_Float16* __restrict__ Q, _Float16* __restrict__ Kt,
                 const float* __restrict__ cosb, const float* __restrict__ sinb)
{
    int idx = blockIdx.x * blockDim.x + threadIdx.x;
    const int total = S_LEN * (NHEAD + NKVH) * 32;
    if (idx >= total) return;
    int j  = idx & 31;
    int hh = (idx >> 5) % (NHEAD + NKVH);
    int s  = idx / (32 * (NHEAD + NKVH));
    float c  = cosb[s * HDIM + j];
    float sn = sinb[s * HDIM + j];
    _Float16* base = (hh < NHEAD)
        ? (Q  + (size_t)s * (NHEAD * HDIM) + hh * HDIM)
        : (Kt + (size_t)s * (NKVH * HDIM) + (hh - NHEAD) * HDIM);
    float a = (float)base[j];
    float b = (float)base[j + 32];
    base[j]      = (_Float16)(a * c - b * sn);
    base[j + 32] = (_Float16)(b * c + a * sn);
}

// ---------------------------------------------------------------------------
// Flash attention, transposed formulation (softmax state per-lane).
// Workgroup: 256 thr (8 waves), q-block 128 (16 rows/wave), k-block 64 via LDS.
// Waves skip compute for k-blocks fully above their causal diagonal.
// ---------------------------------------------------------------------------
__global__ __launch_bounds__(256)
void attn_kernel(const _Float16* __restrict__ Qws,
                 const _Float16* __restrict__ Kws,
                 const _Float16* __restrict__ Vws,
                 const int* __restrict__ special,
                 const int* __restrict__ eos,
                 _Float16* __restrict__ Aws)
{
    __shared__ _Float16 Klds[64 * 64];        // [k_local][d]
    __shared__ _Float16 Vtlds[64 * 64];       // [d][k_local]
    __shared__ unsigned char spec[64];

    const int t    = threadIdx.x;
    const int lane = t & 31;
    const int wave = t >> 5;
    const int nl   = lane & 15;
    const int hi   = lane >> 4;
    const int qb   = blockIdx.x * 128;
    const int h    = blockIdx.y;
    const int kvh  = h / (NHEAD / NKVH);

    const int qrow   = qb + wave * 16 + nl;   // this lane's q column (N index)
    const int qmax   = qb + wave * 16 + 15;   // wave-uniform causal bound
    const int my_eos = eos[qrow];

    // Q^T B-fragments (loop invariant): lane N=q, k-dim = head dim
    F16x16 qf[2];
    #pragma unroll
    for (int dc = 0; dc < 2; ++dc)
        #pragma unroll
        for (int g = 0; g < 8; ++g) {
            int d = dc * 32 + ((g & 4) << 2) + (hi << 3) + ((g & 3) << 1);
            qf[dc].u[g] = *(const unsigned int*)
                &Qws[(size_t)qrow * (NHEAD * HDIM) + h * HDIM + d];
        }

    float m = -1e30f, l = 0.f;
    v8f oacc[4] = {};

    const int kend = qb + 128;
    for (int kb = 0; kb < kend; kb += 64) {
        // ---- stage K row-major, V transposed, special bytes ----
        #pragma unroll
        for (int i = 0; i < 2; ++i) {
            int c    = t + 256 * i;            // 512 chunks of 8 halves
            int krow = c >> 3;
            int d0   = (c & 7) << 3;
            size_t src = (size_t)(kb + krow) * (NKVH * HDIM) + kvh * HDIM + d0;
            uint4 kd = *(const uint4*)&Kws[src];
            *(uint4*)&Klds[krow * 64 + d0] = kd;
            union { uint4 u; _Float16 hv[8]; } vv;
            vv.u = *(const uint4*)&Vws[src];
            #pragma unroll
            for (int j = 0; j < 8; ++j)
                Vtlds[(d0 + j) * 64 + krow] = vv.hv[j];
            if (kb + 64 < kend)  // global_prefetch_b8 for next K block
                __builtin_prefetch(&Kws[src + 64 * (NKVH * HDIM)], 0, 1);
        }
        if (t < 64) spec[t] = (unsigned char)(special[kb + t] != 0);
        __syncthreads();

        if (kb <= qmax) {   // wave-uniform skip of fully-masked blocks
            for (int ks = 0; ks < 64; ks += 32) {
                if (kb + ks > qmax) break;

                // ---- S^T tiles: preload both d-chunk fragments, then WMMA ----
                v8f st[2];
                #pragma unroll
                for (int tile = 0; tile < 2; ++tile) {
                    F16x16 af0, af1;
                    int krow = ks + tile * 16 + nl;
                    #pragma unroll
                    for (int g = 0; g < 8; ++g) {
                        int doff = ((g & 4) << 2) + (hi << 3) + ((g & 3) << 1);
                        af0.u[g] = *(const unsigned int*)&Klds[krow * 64 + doff];
                        af1.u[g] = *(const unsigned int*)&Klds[krow * 64 + 32 + doff];
                    }
                    v8f a = {};
                    a = __builtin_amdgcn_wmma_f32_16x16x32_f16(
                        false, af0.v, false, qf[0].v, (short)0, a, false, false);
                    a = __builtin_amdgcn_wmma_f32_16x16x32_f16(
                        false, af1.v, false, qf[1].v, (short)0, a, false, false);
                    st[tile] = a;
                }

                // ---- mask + online softmax (per-lane column q) ----
                float vals[16];
                #pragma unroll
                for (int tile = 0; tile < 2; ++tile)
                    #pragma unroll
                    for (int r = 0; r < 8; ++r) {
                        int kl   = ks + tile * 16 + r + (hi << 3);
                        int kpos = kb + kl;
                        float sv = st[tile][r] * 0.125f;
                        bool ok  = (kpos <= qrow) && ((kpos >= my_eos) || spec[kl]);
                        vals[tile * 8 + r] = ok ? sv : -1.0e9f;
                    }
                // tree max
                float tmx[16];
                #pragma unroll
                for (int i = 0; i < 16; ++i) tmx[i] = vals[i];
                #pragma unroll
                for (int s2 = 8; s2 >= 1; s2 >>= 1)
                    #pragma unroll
                    for (int i = 0; i < 16; ++i)
                        if (i < s2) tmx[i] = fmaxf(tmx[i], tmx[i + s2]);
                float bmax = fmaxf(tmx[0], __shfl_xor(tmx[0], 16, 32));
                float mnew  = fmaxf(m, bmax);
                float scale = __expf(m - mnew);

                float ps[16];
                _Float16 ph[16];
                #pragma unroll
                for (int i = 0; i < 16; ++i) {
                    ps[i] = __expf(vals[i] - mnew);
                    ph[i] = (_Float16)ps[i];
                }
                #pragma unroll
                for (int s2 = 8; s2 >= 1; s2 >>= 1)
                    #pragma unroll
                    for (int i = 0; i < 16; ++i)
                        if (i < s2) ps[i] += ps[i + s2];
                float lsum = ps[0] + __shfl_xor(ps[0], 16, 32);
                l = l * scale + lsum;
                m = mnew;
                #pragma unroll
                for (int dt = 0; dt < 4; ++dt)
                    #pragma unroll
                    for (int r = 0; r < 8; ++r)
                        oacc[dt][r] *= scale;

                // ---- P^T B-fragment: half-wave exchange via shfl_xor(16) ----
                F16x16 pf;
                #pragma unroll
                for (int g = 0; g < 4; ++g) {
                    unsigned int u0 = pack2(ph[2 * g],     ph[2 * g + 1]);  // tile0 pairs
                    unsigned int u1 = pack2(ph[8 + 2 * g], ph[9 + 2 * g]);  // tile1 pairs
                    unsigned int o0 = __shfl_xor(u0, 16, 32);
                    unsigned int o1 = __shfl_xor(u1, 16, 32);
                    pf.u[g]     = hi ? o1 : u0;
                    pf.u[4 + g] = hi ? u1 : o0;
                }

                // ---- O^T += V^T * P^T : preload all 4 fragments, then WMMA ----
                F16x16 vf[4];
                #pragma unroll
                for (int dt = 0; dt < 4; ++dt) {
                    int d = dt * 16 + nl;
                    #pragma unroll
                    for (int g = 0; g < 8; ++g) {
                        int k = ks + (hi << 4) + (g << 1);
                        vf[dt].u[g] = *(const unsigned int*)&Vtlds[d * 64 + k];
                    }
                }
                #pragma unroll
                for (int dt = 0; dt < 4; ++dt)
                    oacc[dt] = __builtin_amdgcn_wmma_f32_16x16x32_f16(
                        false, vf[dt].v, false, pf.v, (short)0, oacc[dt], false, false);
            }
        }
        __syncthreads();
    }

    // ---- finalize: divide by row sum, vectorized b128 stores ----
    float inv = 1.0f / l;
    #pragma unroll
    for (int dt = 0; dt < 4; ++dt) {
        union { uint4 u; _Float16 hv[8]; } pk;
        #pragma unroll
        for (int r = 0; r < 8; ++r)
            pk.hv[r] = (_Float16)(oacc[dt][r] * inv);
        int d = dt * 16 + (hi << 3);   // r=0..7 -> 8 consecutive halves
        *(uint4*)&Aws[(size_t)qrow * (NHEAD * HDIM) + h * HDIM + d] = pk.u;
    }
}

// ---------------------------------------------------------------------------
extern "C" void kernel_launch(void* const* d_in, const int* in_sizes, int n_in,
                              void* d_out, int out_size, void* d_ws, size_t ws_size,
                              hipStream_t stream) {
    const float* hs      = (const float*)d_in[0];   // [S][D]
    const float* cosb    = (const float*)d_in[1];   // [S][64]
    const float* sinb    = (const float*)d_in[2];
    const float* Wq      = (const float*)d_in[3];   // [D][1024]
    const float* Wk      = (const float*)d_in[4];   // [D][256]
    const float* Wv      = (const float*)d_in[5];
    const float* Wo      = (const float*)d_in[6];   // [1024][D]
    const int*   special = (const int*)d_in[7];     // [S]
    const int*   eos     = (const int*)d_in[8];     // [S]

    _Float16* Qws = (_Float16*)d_ws;                          // S*1024
    _Float16* Kws = Qws + (size_t)S_LEN * DMODEL;             // S*256
    _Float16* Vws = Kws + (size_t)S_LEN * (NKVH * HDIM);      // S*256
    _Float16* Aws = Vws + (size_t)S_LEN * (NKVH * HDIM);      // S*1024

    dim3 blk(256);

    // QKV projections (fp32 in -> f16 ws)
    gemm_f16wmma_kernel<false, true><<<dim3(S_LEN / 128, DMODEL / 128), blk, 0, stream>>>(
        hs, Wq, Qws, S_LEN, DMODEL, DMODEL);
    gemm_f16wmma_kernel<false, true><<<dim3(S_LEN / 128, (NKVH * HDIM) / 128), blk, 0, stream>>>(
        hs, Wk, Kws, S_LEN, NKVH * HDIM, DMODEL);
    gemm_f16wmma_kernel<false, true><<<dim3(S_LEN / 128, (NKVH * HDIM) / 128), blk, 0, stream>>>(
        hs, Wv, Vws, S_LEN, NKVH * HDIM, DMODEL);

    // RoPE on Q and K
    {
        int total = S_LEN * (NHEAD + NKVH) * 32;
        rope_kernel<<<(total + 255) / 256, blk, 0, stream>>>(Qws, Kws, cosb, sinb);
    }

    // Masked flash attention
    attn_kernel<<<dim3(S_LEN / 128, NHEAD), blk, 0, stream>>>(
        Qws, Kws, Vws, special, eos, Aws);

    // Output projection (f16 ws -> fp32 out)
    gemm_f16wmma_kernel<true, false><<<dim3(S_LEN / 128, DMODEL / 128), blk, 0, stream>>>(
        Aws, Wo, (float*)d_out, S_LEN, DMODEL, DMODEL);
}
